// NNUE_48790828482872
// MI455X (gfx1250) — compile-verified, matrix operand order checked by reading, not run on hardware
//
#include <hip/hip_runtime.h>

typedef __attribute__((ext_vector_type(16))) _Float16 v16h;
typedef __attribute__((ext_vector_type(8)))  _Float16 v8h;
typedef __attribute__((ext_vector_type(4)))  _Float16 v4h;
typedef __attribute__((ext_vector_type(8)))  float    v8f;
typedef __attribute__((ext_vector_type(4)))  float    v4f;

#define BATCH   8192
#define NIN     12288
#define H0      256
#define BM      64
#define BK      64
#define BKP     72            // padded row (36 DWORDs) -> conflict-free 16-row frag loads
#define NKITER  (NIN / BK)    // 192

#if defined(__has_builtin)
#  if __has_builtin(__builtin_amdgcn_global_load_async_to_lds_b128)
#    define HAVE_ASYNC_LDS 1
#  else
#    define HAVE_ASYNC_LDS 0
#  endif
#else
#  define HAVE_ASYNC_LDS 0
#endif

union H16 { v16h v; v8h h[2]; };
union A16 { v8h v[2]; _Float16 e[16]; };

__device__ __forceinline__ v8f wmma16x16x32(const v16h a, const v16h b, const v8f c) {
    return __builtin_amdgcn_wmma_f32_16x16x32_f16(false, a, false, b, (short)0, c, false, false);
}

#if HAVE_ASYNC_LDS
// Builtin signature (from clang diagnostic): param0 = int4 (AS1 "__device__") pointer,
// param1 = LDS-side pointer, then imm offset, imm cpol.
typedef int v4i_g __attribute__((__vector_size__(4 * sizeof(int))));
typedef __attribute__((address_space(1))) v4i_g* gasync_t;
typedef __attribute__((address_space(3))) v4i_g* lasync_t;

__device__ __forceinline__ gasync_t to_global(const void* p) {
    return (gasync_t)(unsigned long long)(__SIZE_TYPE__)p;
}
__device__ __forceinline__ lasync_t to_lds(void* p) {
    return (lasync_t)(unsigned int)(__SIZE_TYPE__)p;   // generic LDS addr low 32 bits = LDS offset
}
#endif

__device__ __forceinline__ void wait_async0() {
#if defined(__has_builtin) && __has_builtin(__builtin_amdgcn_s_wait_asynccnt)
    __builtin_amdgcn_s_wait_asynccnt(0);
#else
    asm volatile("s_wait_asynccnt 0x0" ::: "memory");
#endif
}

// ---------------------------------------------------------------------------
// f32 -> f16 weight conversion (vectorized x4)
// ---------------------------------------------------------------------------
__global__ __launch_bounds__(256) void cvt_f32_f16(const float* __restrict__ src,
                                                   _Float16* __restrict__ dst, int n) {
    int i = (blockIdx.x * 256 + threadIdx.x) * 4;
    if (i < n) {
        v4f s = *(const v4f*)(src + i);
        v4h d;
        d[0] = (_Float16)s[0]; d[1] = (_Float16)s[1];
        d[2] = (_Float16)s[2]; d[3] = (_Float16)s[3];
        *(v4h*)(dst + i) = d;
    }
}

// ---------------------------------------------------------------------------
// Feature-transformer GEMM:  o0[b, 512] = clip(feat @ ft_w^T + ft_b, 0, 1)
// grid.x = 128 (M blocks of 64), grid.y = side (0=white GEMM, 1=black GEMM)
// ---------------------------------------------------------------------------
__global__ __launch_bounds__(256) void ft_gemm(const float* __restrict__ wfeat,
                                               const float* __restrict__ bfeat,
                                               const unsigned char* __restrict__ side,
                                               const _Float16* __restrict__ W16,
                                               const float* __restrict__ ftb,
                                               _Float16* __restrict__ o0) {
    __shared__ alignas(16) _Float16 As[2][BM][BKP];   // 18 KB
    __shared__ alignas(16) _Float16 Bs[2][H0][BKP];   // 72 KB

    const int tid  = threadIdx.x;
    const int lane = tid & 31;
    const int wave = tid >> 5;
    const int wm   = wave >> 2;          // 0..1  (32 rows each)
    const int wn   = wave & 3;           // 0..3  (64 cols each)
    const int lm   = lane & 15;
    const int hi   = (lane >> 4) & 1;
    const int m0   = blockIdx.x * BM;
    const bool isWhite = (blockIdx.y == 0);
    const float* A = isWhite ? wfeat : bfeat;

    // cooperative staging roles
    const int arow = tid >> 2;           // 0..63, 16 f32 each
    const int acol = (tid & 3) * 16;
    const int brow = tid;                // 0..255, 64 f16 each

    v8f acc[2][4];
    #pragma unroll
    for (int mt = 0; mt < 2; ++mt)
        #pragma unroll
        for (int nt = 0; nt < 4; ++nt) acc[mt][nt] = v8f{};

    v4f aReg[4];
#if !HAVE_ASYNC_LDS
    uint4 bReg[8];
#endif

    // ---- preload k-chunk 0 ----
    {
        const float* ap = A + (size_t)(m0 + arow) * NIN + acol;
        #pragma unroll
        for (int j = 0; j < 4; ++j) aReg[j] = *(const v4f*)(ap + j * 4);
#if HAVE_ASYNC_LDS
        const _Float16* gsrc = W16 + (size_t)brow * NIN;
        _Float16*       ldst = &Bs[0][brow][0];
        #pragma unroll
        for (int j = 0; j < 8; ++j)
            __builtin_amdgcn_global_load_async_to_lds_b128(
                to_global(gsrc + j * 8), to_lds(ldst + j * 8), 0, 0);
#else
        const _Float16* bp = W16 + (size_t)brow * NIN;
        #pragma unroll
        for (int j = 0; j < 8; ++j) bReg[j] = *(const uint4*)(bp + j * 8);
#endif
    }

    for (int ko = 0; ko < NKITER; ++ko) {
        const int buf = ko & 1;
        // A regs -> LDS (f32 -> f16 conversion here)
        {
            A16 at;
            #pragma unroll
            for (int j = 0; j < 4; ++j) {
                at.e[4 * j + 0] = (_Float16)aReg[j][0];
                at.e[4 * j + 1] = (_Float16)aReg[j][1];
                at.e[4 * j + 2] = (_Float16)aReg[j][2];
                at.e[4 * j + 3] = (_Float16)aReg[j][3];
            }
            *(v8h*)&As[buf][arow][acol]     = at.v[0];
            *(v8h*)&As[buf][arow][acol + 8] = at.v[1];
#if !HAVE_ASYNC_LDS
            #pragma unroll
            for (int j = 0; j < 8; ++j)
                *(uint4*)&Bs[buf][brow][j * 8] = bReg[j];
#endif
        }
#if HAVE_ASYNC_LDS
        wait_async0();                   // this wave's B tile for `buf` is in LDS
#endif
        __syncthreads();                 // everyone's A+B for `buf` visible; buf^1 free
        // prefetch next k-chunk (A into regs; B async straight into LDS buf^1)
        if (ko + 1 < NKITER) {
            const int k0 = (ko + 1) * BK;
            const float* ap = A + (size_t)(m0 + arow) * NIN + k0 + acol;
            #pragma unroll
            for (int j = 0; j < 4; ++j) aReg[j] = *(const v4f*)(ap + j * 4);
#if HAVE_ASYNC_LDS
            const _Float16* gsrc = W16 + (size_t)brow * NIN + k0;
            _Float16*       ldst = &Bs[buf ^ 1][brow][0];
            #pragma unroll
            for (int j = 0; j < 8; ++j)
                __builtin_amdgcn_global_load_async_to_lds_b128(
                    to_global(gsrc + j * 8), to_lds(ldst + j * 8), 0, 0);
#else
            const _Float16* bp = W16 + (size_t)brow * NIN + k0;
            #pragma unroll
            for (int j = 0; j < 8; ++j) bReg[j] = *(const uint4*)(bp + j * 8);
#endif
        }
        // compute: 2 K-steps of 32 on this buffer
        #pragma unroll
        for (int ks = 0; ks < 2; ++ks) {
            const int kb = ks * 32 + hi * 8;
            H16 afr[2];
            #pragma unroll
            for (int mt = 0; mt < 2; ++mt) {
                const int r = wm * 32 + mt * 16 + lm;
                afr[mt].h[0] = *(const v8h*)&As[buf][r][kb];
                afr[mt].h[1] = *(const v8h*)&As[buf][r][kb + 16];
            }
            #pragma unroll
            for (int nt = 0; nt < 4; ++nt) {
                const int n = wn * 64 + nt * 16 + lm;
                H16 bfr;
                bfr.h[0] = *(const v8h*)&Bs[buf][n][kb];
                bfr.h[1] = *(const v8h*)&Bs[buf][n][kb + 16];
                #pragma unroll
                for (int mt = 0; mt < 2; ++mt)
                    acc[mt][nt] = wmma16x16x32(afr[mt].v, bfr.v, acc[mt][nt]);
            }
        }
    }

    // epilogue: bias, clip[0,1], side-dependent concat, f16 store
    float bias[4];
    #pragma unroll
    for (int nt = 0; nt < 4; ++nt) bias[nt] = ftb[wn * 64 + nt * 16 + lm];

    #pragma unroll
    for (int mt = 0; mt < 2; ++mt) {
        #pragma unroll
        for (int i = 0; i < 8; ++i) {
            const int row = m0 + wm * 32 + mt * 16 + i + hi * 8;
            const bool s = side[row] != 0;
            const int half = (s == isWhite) ? 0 : 256;   // side? [w,b] : [b,w]
            _Float16* orow = o0 + (size_t)row * 512 + half;
            #pragma unroll
            for (int nt = 0; nt < 4; ++nt) {
                float v = acc[mt][nt][i] + bias[nt];
                v = fminf(fmaxf(v, 0.0f), 1.0f);
                orow[wn * 64 + nt * 16 + lm] = (_Float16)v;
            }
        }
    }
}

// ---------------------------------------------------------------------------
// Head: L1 (512->32) + L2 (32->32) + L3 (32->1) + sigmoid.
// One wave per 16 batch rows; 8 waves per block; grid = 64.
// ---------------------------------------------------------------------------
__global__ __launch_bounds__(256) void head(const _Float16* __restrict__ o0,
                                            const _Float16* __restrict__ l1w,
                                            const float* __restrict__ l1b,
                                            const _Float16* __restrict__ l2w,
                                            const float* __restrict__ l2b,
                                            const float* __restrict__ l3w,
                                            const float* __restrict__ l3b,
                                            float* __restrict__ out) {
    __shared__ alignas(16) _Float16 hbuf[8][16][40];   // padded wave-private C->A relayout

    const int tid  = threadIdx.x;
    const int lane = tid & 31;
    const int wave = tid >> 5;
    const int lm   = lane & 15;
    const int hi   = (lane >> 4) & 1;
    const int r0   = blockIdx.x * 128 + wave * 16;

    // ---- L1: [16 x 512] @ [512 x 32] ----
    v8f acc1[2] = {v8f{}, v8f{}};
    const _Float16* arow = o0 + (size_t)(r0 + lm) * 512;
    #pragma unroll
    for (int ks = 0; ks < 16; ++ks) {
        const int kb = ks * 32 + hi * 8;
        H16 a;
        a.h[0] = *(const v8h*)(arow + kb);
        a.h[1] = *(const v8h*)(arow + kb + 16);
        #pragma unroll
        for (int nt = 0; nt < 2; ++nt) {
            const _Float16* bp = l1w + (size_t)(nt * 16 + lm) * 512;
            H16 b;
            b.h[0] = *(const v8h*)(bp + kb);
            b.h[1] = *(const v8h*)(bp + kb + 16);
            acc1[nt] = wmma16x16x32(a.v, b.v, acc1[nt]);
        }
    }
    #pragma unroll
    for (int nt = 0; nt < 2; ++nt) {
        const int col = nt * 16 + lm;
        const float bv = l1b[col];
        #pragma unroll
        for (int i = 0; i < 8; ++i) {
            float v = fminf(fmaxf(acc1[nt][i] + bv, 0.0f), 1.0f);
            hbuf[wave][i + hi * 8][col] = (_Float16)v;
        }
    }

    // ---- L2: [16 x 32] @ [32 x 32], single K-step ----
    {
        const int kb = hi * 8;
        H16 a;
        a.h[0] = *(const v8h*)&hbuf[wave][lm][kb];
        a.h[1] = *(const v8h*)&hbuf[wave][lm][kb + 16];
        v8f acc2[2] = {v8f{}, v8f{}};
        #pragma unroll
        for (int nt = 0; nt < 2; ++nt) {
            const _Float16* bp = l2w + (size_t)(nt * 16 + lm) * 32;
            H16 b;
            b.h[0] = *(const v8h*)(bp + kb);
            b.h[1] = *(const v8h*)(bp + kb + 16);
            acc2[nt] = wmma16x16x32(a.v, b.v, acc2[nt]);
        }
        #pragma unroll
        for (int nt = 0; nt < 2; ++nt) {
            const int col = nt * 16 + lm;
            const float bv = l2b[col];
            #pragma unroll
            for (int i = 0; i < 8; ++i) {
                float v = fminf(fmaxf(acc2[nt][i] + bv, 0.0f), 1.0f);
                hbuf[wave][i + hi * 8][col] = (_Float16)v;
            }
        }
    }

    // ---- L3 + sigmoid on lanes 0..15 ----
    if (lane < 16) {
        float dot = 0.0f;
        #pragma unroll
        for (int k = 0; k < 32; ++k)
            dot += (float)hbuf[wave][lane][k] * l3w[k];
        const float o3 = (dot + l3b[0]) * 300.0f;
        out[r0 + lane] = 1.0f / (1.0f + __expf(-o3 * (1.0f / 200.0f)));
    }
}

// ---------------------------------------------------------------------------
extern "C" void kernel_launch(void* const* d_in, const int* in_sizes, int n_in,
                              void* d_out, int out_size, void* d_ws, size_t ws_size,
                              hipStream_t stream) {
    const float*         wfeat = (const float*)d_in[0];
    const float*         bfeat = (const float*)d_in[1];
    const unsigned char* side  = (const unsigned char*)d_in[2];
    const float*         ft_w  = (const float*)d_in[3];
    const float*         ft_b  = (const float*)d_in[4];
    const float*         l1_w  = (const float*)d_in[5];
    const float*         l1_b  = (const float*)d_in[6];
    const float*         l2_w  = (const float*)d_in[7];
    const float*         l2_b  = (const float*)d_in[8];
    const float*         l3_w  = (const float*)d_in[9];
    const float*         l3_b  = (const float*)d_in[10];
    float*               out   = (float*)d_out;

    char* ws = (char*)d_ws;
    const size_t NFT = (size_t)H0 * NIN;          // 3,145,728
    _Float16* W16   = (_Float16*)ws;                              // 6,291,456 B
    _Float16* L1W16 = (_Float16*)(ws + 6291456);                  //    32,768 B
    _Float16* L2W16 = (_Float16*)(ws + 6291456 + 32768);          //     2,048 B
    _Float16* O0    = (_Float16*)(ws + 6291456 + 32768 + 2048);   // 8,388,608 B

    cvt_f32_f16<<<(int)(NFT / 4 / 256), 256, 0, stream>>>(ft_w, W16, (int)NFT);
    cvt_f32_f16<<<16, 256, 0, stream>>>(l1_w, L1W16, 32 * 512);
    cvt_f32_f16<<<1, 256, 0, stream>>>(l2_w, L2W16, 32 * 32);

    ft_gemm<<<dim3(BATCH / BM, 2), 256, 0, stream>>>(wfeat, bfeat, side, W16, ft_b, O0);

    head<<<BATCH / 128, 256, 0, stream>>>(O0, L1W16, l1_b, L2W16, l2_b, l3_w, l3_b, out);
}
